// AttentionModule_71674414235796
// MI455X (gfx1250) — compile-verified, matrix operand order checked by reading, not run on hardware
//
#include <hip/hip_runtime.h>

#define QUERY_DIM   256
#define NUM_CLASSES 5000
#define SEQ_LEN     2048
#define BATCH       16

#define THREADS 256    // 8 waves: allows >256 VGPRs/wave -> no scratch spills
#define NWAVES  8
#define LBLK    32     // L-block streamed per iteration
#define CTILE   128    // classes per workgroup (8 waves x 16)
#define KROW    264    // keys_lds row stride in halves (256 + 8 skew)
#define KTROW   40     // keysT_lds row stride in halves (32 + 8 skew)
#define PROW    40     // P_lds row stride in halves

typedef __attribute__((ext_vector_type(16))) __bf16 v16bf;
typedef __attribute__((ext_vector_type(8)))  __bf16 v8bf;
typedef __attribute__((ext_vector_type(4)))  __bf16 v4bf;
typedef __attribute__((ext_vector_type(8)))  float  v8f;
typedef __attribute__((ext_vector_type(4)))  float  v4f;

__global__ __launch_bounds__(THREADS)
void attn_pool_kernel(const float* __restrict__ keys,
                      const float* __restrict__ U,
                      float* __restrict__ out) {
  __shared__ __bf16 keys_lds [LBLK * KROW];          // row-major keys block (bf16)
  __shared__ __bf16 keysT_lds[QUERY_DIM * KTROW];    // transposed keys block (bf16)
  __shared__ __bf16 P_lds    [NWAVES * 16 * PROW];   // per-wave softmax weights

  const int t    = threadIdx.x;
  const int lane = t & 31;
  const int w    = t >> 5;        // wave id 0..7
  const int ln   = lane & 15;     // lane within 16-group
  const int hi   = lane >> 4;     // 0 or 1 (K-half selector in A/B frags)

  const int b     = blockIdx.y;
  const int cbase = blockIdx.x * CTILE + w * 16;     // this wave's 16 classes

  // ---- Load U tile as bf16 A-fragments, pre-scaled by 1/sqrt(256)=1/16 (exact) ----
  v16bf a_u[8];
  {
    int row = cbase + ln;
    if (row >= NUM_CLASSES) row = NUM_CLASSES - 1;   // clamp; store is guarded
    const float* Ur = U + (size_t)row * QUERY_DIM;
    #pragma unroll
    for (int kc = 0; kc < 8; ++kc) {
      const int k0 = kc * 32 + hi * 8;
      #pragma unroll
      for (int i = 0; i < 8; ++i) {
        a_u[kc][i]     = (__bf16)(Ur[k0 + i]      * 0.0625f);
        a_u[kc][i + 8] = (__bf16)(Ur[k0 + 16 + i] * 0.0625f);
      }
    }
  }

  // ---- Online-softmax state: per lane, element j <-> class row (j + 8*hi) ----
  v8f acc[16];                 // 16 d-tiles x 8 rows of f32 accumulator (16x256)
  v8f m_run, l_run;
  #pragma unroll
  for (int i = 0; i < 16; ++i)
    #pragma unroll
    for (int j = 0; j < 8; ++j) acc[i][j] = 0.0f;
  #pragma unroll
  for (int j = 0; j < 8; ++j) { m_run[j] = -1.0e30f; l_run[j] = 0.0f; }

  __bf16* Pw = &P_lds[w * 16 * PROW];

  for (int it = 0; it < SEQ_LEN / LBLK; ++it) {
    const int l0 = it * LBLK;
    __syncthreads();   // previous iteration's LDS consumers done

    // ---- Stage keys block row-major: thread -> (row r, 32-wide d-slab) ----
    {
      const int r  = t >> 3;              // 0..31
      const int d0 = (t & 7) * 32;        // 0..224
      const float* src = keys + (((size_t)b * SEQ_LEN + l0 + r) * QUERY_DIM + d0);
      __bf16* dst = &keys_lds[r * KROW + d0];
      #pragma unroll
      for (int i = 0; i < 8; ++i) {
        v4f v = *(const v4f*)(src + i * 4);
        v4bf o;
        o[0] = (__bf16)v[0]; o[1] = (__bf16)v[1];
        o[2] = (__bf16)v[2]; o[3] = (__bf16)v[3];
        *(v4bf*)(dst + i * 4) = o;
      }
      // Prefetch next L-block's slab (one cacheline/thread covers the block).
      if (it + 1 < SEQ_LEN / LBLK)
        __builtin_prefetch(src + LBLK * QUERY_DIM, 0, 3);
    }
    // ---- Stage keys block transposed: thread -> one d column (coalesced reads) ----
    {
      const float* src = keys + (((size_t)b * SEQ_LEN + l0) * QUERY_DIM + t);
      __bf16* dst = &keysT_lds[t * KTROW];
      #pragma unroll
      for (int r = 0; r < LBLK; ++r)
        dst[r] = (__bf16)src[(size_t)r * QUERY_DIM];
    }
    __syncthreads();

    // ---- S = (U/16) * K^T : M=16 classes, N=32 l, K=256 d ----
    v8f s[2];
    #pragma unroll
    for (int nt = 0; nt < 2; ++nt) {
      v8f c;
      #pragma unroll
      for (int j = 0; j < 8; ++j) c[j] = 0.0f;
      #pragma unroll
      for (int kc = 0; kc < 8; ++kc) {
        const __bf16* p = &keys_lds[(nt * 16 + ln) * KROW + kc * 32 + hi * 8];
        v8bf blo = *(const v8bf*)p;
        v8bf bh  = *(const v8bf*)(p + 16);
        v16bf bb;
        #pragma unroll
        for (int i = 0; i < 8; ++i) { bb[i] = blo[i]; bb[i + 8] = bh[i]; }
        c = __builtin_amdgcn_wmma_f32_16x16x32_bf16(false, a_u[kc], false, bb,
                                                    (short)0, c, false, false);
      }
      s[nt] = c;
    }

    // ---- Online softmax over this L-block (rows live across 16-lane halves) ----
    v8f rmax;
    #pragma unroll
    for (int j = 0; j < 8; ++j) rmax[j] = fmaxf(s[0][j], s[1][j]);
    #pragma unroll
    for (int off = 1; off < 16; off <<= 1)
      #pragma unroll
      for (int j = 0; j < 8; ++j)
        rmax[j] = fmaxf(rmax[j], __shfl_xor(rmax[j], off, 32));

    v8f mnew, p0, p1, rsum, corr;
    #pragma unroll
    for (int j = 0; j < 8; ++j) {
      mnew[j] = fmaxf(m_run[j], rmax[j]);
      p0[j]   = __expf(s[0][j] - mnew[j]);
      p1[j]   = __expf(s[1][j] - mnew[j]);
      rsum[j] = p0[j] + p1[j];
      corr[j] = __expf(m_run[j] - mnew[j]);
    }
    #pragma unroll
    for (int off = 1; off < 16; off <<= 1)
      #pragma unroll
      for (int j = 0; j < 8; ++j)
        rsum[j] += __shfl_xor(rsum[j], off, 32);
    #pragma unroll
    for (int j = 0; j < 8; ++j) {
      l_run[j] = l_run[j] * corr[j] + rsum[j];
      m_run[j] = mnew[j];
    }
    #pragma unroll
    for (int dt = 0; dt < 16; ++dt)
      #pragma unroll
      for (int j = 0; j < 8; ++j) acc[dt][j] *= corr[j];

    // ---- P: C-layout -> LDS -> A-layout (per-wave region; same-wave DS ordering) ----
    #pragma unroll
    for (int j = 0; j < 8; ++j) {
      Pw[(j + 8 * hi) * PROW + ln]      = (__bf16)p0[j];
      Pw[(j + 8 * hi) * PROW + 16 + ln] = (__bf16)p1[j];
    }
    v16bf pa;
    {
      const __bf16* pp = &Pw[ln * PROW + hi * 8];
      v8bf alo = *(const v8bf*)pp;
      v8bf ah  = *(const v8bf*)(pp + 16);
      #pragma unroll
      for (int i = 0; i < 8; ++i) { pa[i] = alo[i]; pa[i + 8] = ah[i]; }
    }

    // ---- acc += P * K : M=16 classes, N=256 d, K=32 l ----
    #pragma unroll
    for (int dt = 0; dt < 16; ++dt) {
      const __bf16* q = &keysT_lds[(dt * 16 + ln) * KTROW + hi * 8];
      v8bf blo = *(const v8bf*)q;
      v8bf bh  = *(const v8bf*)(q + 16);
      v16bf bb;
      #pragma unroll
      for (int i = 0; i < 8; ++i) { bb[i] = blo[i]; bb[i + 8] = bh[i]; }
      acc[dt] = __builtin_amdgcn_wmma_f32_16x16x32_bf16(false, pa, false, bb,
                                                        (short)0, acc[dt], false, false);
    }
  }

  // ---- Epilogue: v = acc / l, scatter to out[b][c][d] ----
  v8f inv;
  #pragma unroll
  for (int j = 0; j < 8; ++j) inv[j] = 1.0f / l_run[j];
  #pragma unroll
  for (int j = 0; j < 8; ++j) {
    const int c = cbase + j + 8 * hi;
    if (c < NUM_CLASSES) {
      float* dst = out + ((size_t)b * NUM_CLASSES + c) * QUERY_DIM;
      #pragma unroll
      for (int dt = 0; dt < 16; ++dt)
        dst[dt * 16 + ln] = acc[dt][j] * inv[j];
    }
  }
}

extern "C" void kernel_launch(void* const* d_in, const int* in_sizes, int n_in,
                              void* d_out, int out_size, void* d_ws, size_t ws_size,
                              hipStream_t stream) {
  const float* keys = (const float*)d_in[0];   // (16, 2048, 256) f32
  const float* U    = (const float*)d_in[1];   // (5000, 256) f32
  float* out        = (float*)d_out;           // (16, 5000, 256) f32

  dim3 grid((NUM_CLASSES + CTILE - 1) / CTILE, BATCH);  // 40 x 16
  attn_pool_kernel<<<grid, THREADS, 0, stream>>>(keys, U, out);
}
